// FairFightConvQAT_59476707115413
// MI455X (gfx1250) — 8-bit matrix operand order repaired and checked by reading, compiled, not run on hardware
//
#include <hip/hip_runtime.h>


typedef int      v8i  __attribute__((ext_vector_type(8)));
typedef _Float16 h8   __attribute__((ext_vector_type(8)));

#define NBATCH 256
#define INV127 (1.0f / 127.0f)

#define OFF_HDR   0
#define OFF_W1Q   256
#define OFF_W4Q   512
#define OFF_W2P   1024
#define OFF_W3P   4096
#define OFF_A1    8192
#define SZ_A1     ((size_t)NBATCH*126*126*16*2)
#define OFF_A2    (OFF_A1 + SZ_A1)
#define SZ_A2     ((size_t)NBATCH*124*124*16*2)
#define OFF_A3    (OFF_A2 + SZ_A2)

__device__ __forceinline__ int q8(float x) {
  int q = (int)rintf(x);
  return q > 127 ? 127 : (q < -128 ? -128 : q);
}

__global__ void qat_init(unsigned* hdr) {
  if (threadIdx.x < 64) hdr[threadIdx.x] = 0u;
}

__global__ void __launch_bounds__(256)
qat_absmax(const float* __restrict__ p, long long nElems, unsigned* __restrict__ slot) {
  __shared__ float red[256];
  const int tid = threadIdx.x;
  float m = 0.0f;
  for (long long i = (long long)blockIdx.x * 256 + tid; i < nElems;
       i += (long long)gridDim.x * 256)
    m = fmaxf(m, fabsf(p[i]));
  red[tid] = m; __syncthreads();
  for (int s = 128; s > 0; s >>= 1) {
    if (tid < s) red[tid] = fmaxf(red[tid], red[tid + s]);
    __syncthreads();
  }
  if (tid == 0) atomicMax(slot, __float_as_uint(red[0]));
}

__global__ void __launch_bounds__(256)
qat_prep(const float* __restrict__ w1, const float* __restrict__ w2,
         const float* __restrict__ w3, const float* __restrict__ w4,
         unsigned* __restrict__ hdr, signed char* __restrict__ w1q,
         int* __restrict__ w2p, int* __restrict__ w3p, signed char* __restrict__ w4q) {
  __shared__ float red[256];
  const int tid = threadIdx.x;
  auto absmaxOf = [&](const float* p, int nE) -> float {
    float m = 0.0f;
    for (int i = tid; i < nE; i += 256) m = fmaxf(m, fabsf(p[i]));
    red[tid] = m; __syncthreads();
    for (int s = 128; s > 0; s >>= 1) {
      if (tid < s) red[tid] = fmaxf(red[tid], red[tid + s]);
      __syncthreads();
    }
    const float r = red[0]; __syncthreads();
    return r;
  };
  const float m1 = fmaxf(absmaxOf(w1, 144),  1e-8f);
  const float m2 = fmaxf(absmaxOf(w2, 2304), 1e-8f);
  const float m3 = fmaxf(absmaxOf(w3, 2304), 1e-8f);
  const float m4 = fmaxf(absmaxOf(w4, 64),   1e-8f);
  const float i1 = 127.0f * __builtin_amdgcn_rcpf(m1);
  const float i2 = 127.0f * __builtin_amdgcn_rcpf(m2);
  const float i3 = 127.0f * __builtin_amdgcn_rcpf(m3);
  const float i4 = 127.0f * __builtin_amdgcn_rcpf(m4);
  if (tid == 0) {
    ((float*)hdr)[8]  = m1 * INV127; ((float*)hdr)[9]  = m2 * INV127;
    ((float*)hdr)[10] = m3 * INV127; ((float*)hdr)[11] = m4 * INV127;
  }
  for (int i = tid; i < 144; i += 256)
    w1q[i] = (signed char)q8(w1[i] * i1);
  for (int i = tid; i < 64; i += 256) {
    const int tap = i >> 4, cin = i & 15;
    w4q[i] = (signed char)q8(w4[cin * 4 + tap] * i4);
  }
  auto pack3x3 = [&](const float* w, float inv, int* dst) {
    for (int idx = tid; idx < 768; idx += 256) {
      const int ch = idx >> 8;
      const int ln = (idx >> 3) & 31;
      const int v  = idx & 7;
      const int m = ln & 15, hf = ln >> 4;
      unsigned word = 0;
      #pragma unroll
      for (int b = 0; b < 4; ++b) {
        const int k = ch * 64 + (v >> 2) * 32 + hf * 16 + (v & 3) * 4 + b;
        int val = 0;
        if (k < 144) {
          const int tap = k >> 4, cin = k & 15;
          val = q8(w[(m * 16 + cin) * 9 + tap] * inv);
        }
        word |= (unsigned)(val & 255) << (b * 8);
      }
      dst[idx] = word;
    }
  };
  pack3x3(w2, i2, w2p);
  pack3x3(w3, i3, w3p);
}

__global__ void __launch_bounds__(256)
qat_conv1(const float* __restrict__ x, _Float16* __restrict__ out,
          const signed char* __restrict__ w1q, const float* __restrict__ b1,
          unsigned* __restrict__ hdr) {
  __shared__ float red[256];
  const int tid = threadIdx.x;
  const long long pix   = (long long)blockIdx.x * 256 + tid;
  const long long total = (long long)NBATCH * 126 * 126;
  float lmax = 0.0f;
  const float xMax = fmaxf(__uint_as_float(hdr[0]), 1e-8f);
  const float invS = 127.0f * __builtin_amdgcn_rcpf(xMax);
  const float s    = xMax * __uint_as_float(hdr[8]) * INV127;
  if (pix < total) {
    const int n   = (int)(pix / (126 * 126));
    const int rem = (int)(pix % (126 * 126));
    const int r = rem / 126, c = rem % 126;
    int xq[9];
    #pragma unroll
    for (int t = 0; t < 9; ++t) {
      const int dy = t / 3, dx = t % 3;
      xq[t] = q8(x[(size_t)n * 16384 + (size_t)(r + dy) * 128 + (c + dx)] * invS);
    }
    union { _Float16 h[16]; uint4 u[2]; } pk;
    #pragma unroll
    for (int m = 0; m < 16; ++m) {
      int acc = 0;
      #pragma unroll
      for (int t = 0; t < 9; ++t) acc += xq[t] * (int)w1q[m * 9 + t];
      float y = (float)acc * s + b1[m];
      y = fmaxf(y, 0.0f);
      pk.h[m] = (_Float16)y;
      lmax = fmaxf(lmax, y);
    }
    _Float16* dst = out + (size_t)pix * 16;
    *(uint4*)(dst)     = pk.u[0];
    *(uint4*)(dst + 8) = pk.u[1];
  }
  red[tid] = lmax; __syncthreads();
  for (int s2 = 128; s2 > 0; s2 >>= 1) {
    if (tid < s2) red[tid] = fmaxf(red[tid], red[tid + s2]);
    __syncthreads();
  }
  if (tid == 0) atomicMax(&hdr[1], __float_as_uint(red[0]));
}

__global__ void __launch_bounds__(256)
qat_conv3x3_i8wmma(const _Float16* __restrict__ in, _Float16* __restrict__ out,
                   const int* __restrict__ packW, const float* __restrict__ bias,
                   unsigned* __restrict__ hdr, int inMaxSlot, int wScaleSlot,
                   int outMaxSlot, int Hin, int Win, int Hout, int Wout,
                   int nColT, int nRowT) {
  __shared__ __align__(16) unsigned char tileQ[18 * 18 * 16];
  __shared__ float red[256];
  const int tid  = threadIdx.x;
  const int lane = tid & 31;
  const int wave = tid >> 5;
  int bId = blockIdx.x;
  const int ct = bId % nColT; bId /= nColT;
  const int rt = bId % nRowT; bId /= nRowT;
  const int n  = bId;
  const int R0 = rt * 16;
  const int C0 = ct * 16;

  const float inMax   = fmaxf(__uint_as_float(hdr[inMaxSlot]), 1e-8f);
  const float invS    = 127.0f * __builtin_amdgcn_rcpf(inMax);
  const float s       = inMax * __uint_as_float(hdr[wScaleSlot]) * INV127;

  for (int p = tid; p < 324; p += 256) {
    const int rr = p / 18, cc = p % 18;
    const int gr = R0 + rr, gc = C0 + cc;
    uint4 q = make_uint4(0u, 0u, 0u, 0u);
    if (gr < Hin && gc < Win) {
      const _Float16* src = in + (((size_t)n * Hin + gr) * Win + gc) * 16;
      const h8 v0 = *(const h8*)(src);
      const h8 v1 = *(const h8*)(src + 8);
      unsigned wds[4] = {0u, 0u, 0u, 0u};
      #pragma unroll
      for (int c = 0; c < 8; ++c) {
        const int qa = q8((float)v0[c] * invS);
        const int qb = q8((float)v1[c] * invS);
        wds[c >> 2]       |= (unsigned)(qa & 255) << ((c & 3) * 8);
        wds[(c >> 2) + 2] |= (unsigned)(qb & 255) << ((c & 3) * 8);
      }
      q = make_uint4(wds[0], wds[1], wds[2], wds[3]);
    }
    *(uint4*)(tileQ + p * 16) = q;
  }
  __syncthreads();

  const v8i* pa = (const v8i*)packW;
  const v8i a0 = pa[0 * 32 + lane];
  const v8i a1 = pa[1 * 32 + lane];
  const v8i a2 = pa[2 * 32 + lane];

  const int col  = lane & 15;
  const int half = lane >> 4;

  v8i acc[2] = {{0,0,0,0,0,0,0,0}, {0,0,0,0,0,0,0,0}};
  #pragma unroll
  for (int sr = 0; sr < 2; ++sr) {
    const int baseRow = wave * 2 + sr;
    #pragma unroll
    for (int ch = 0; ch < 3; ++ch) {
      const int tapLo = ch * 4 + half;
      const int tapHi = ch * 4 + 2 + half;
      uint4 blo = make_uint4(0u, 0u, 0u, 0u), bhi = make_uint4(0u, 0u, 0u, 0u);
      if (tapLo < 9) {
        const int dy = tapLo / 3, dx = tapLo % 3;
        blo = *(const uint4*)(tileQ + ((baseRow + dy) * 18 + (col + dx)) * 16);
      }
      if (tapHi < 9) {
        const int dy = tapHi / 3, dx = tapHi % 3;
        bhi = *(const uint4*)(tileQ + ((baseRow + dy) * 18 + (col + dx)) * 16);
      }
      v8i bv;
      bv[0] = (int)blo.x; bv[1] = (int)blo.y; bv[2] = (int)blo.z; bv[3] = (int)blo.w;
      bv[4] = (int)bhi.x; bv[5] = (int)bhi.y; bv[6] = (int)bhi.z; bv[7] = (int)bhi.w;
      const v8i av = (ch == 0) ? a0 : (ch == 1) ? a1 : a2;
      acc[sr] = __builtin_amdgcn_wmma_i32_16x16x64_iu8(true, av, true, bv, acc[sr], false, false);
    }
  }

  const int   cc       = C0 + col;
  const int   coutBase = half * 8;
  float lmax = 0.0f;
  #pragma unroll
  for (int sr = 0; sr < 2; ++sr) {
    const int r = R0 + wave * 2 + sr;
    union { _Float16 h[8]; uint4 u; } pk;
    float rmax = 0.0f;
    #pragma unroll
    for (int j = 0; j < 8; ++j) {
      float y = (float)acc[sr][j] * s + bias[coutBase + j];
      y = fmaxf(y, 0.0f);
      pk.h[j] = (_Float16)y;
      rmax = fmaxf(rmax, y);
    }
    if (r < Hout && cc < Wout) {
      *(uint4*)(out + (((size_t)n * Hout + r) * Wout + cc) * 16 + coutBase) = pk.u;
      lmax = fmaxf(lmax, rmax);
    }
  }

  red[tid] = lmax; __syncthreads();
  for (int s2 = 128; s2 > 0; s2 >>= 1) {
    if (tid < s2) red[tid] = fmaxf(red[tid], red[tid + s2]);
    __syncthreads();
  }
  if (tid == 0) atomicMax(&hdr[outMaxSlot], __float_as_uint(red[0]));
}

__global__ void __launch_bounds__(256)
qat_conv4(const _Float16* __restrict__ in, float* __restrict__ out,
          const signed char* __restrict__ w4q, const float* __restrict__ b4,
          const unsigned* __restrict__ hdr) {
  const long long pix   = (long long)blockIdx.x * 256 + threadIdx.x;
  const long long total = (long long)NBATCH * 121 * 121;
  if (pix >= total) return;
  const int n   = (int)(pix / (121 * 121));
  const int rem = (int)(pix % (121 * 121));
  const int r = rem / 121, c = rem % 121;
  const float inMax = fmaxf(__uint_as_float(hdr[3]), 1e-8f);
  const float invS  = 127.0f * __builtin_amdgcn_rcpf(inMax);
  const float s     = inMax * __uint_as_float(hdr[11]) * INV127;
  int acc = 0;
  #pragma unroll
  for (int t = 0; t < 4; ++t) {
    const int dy = t >> 1, dx = t & 1;
    const _Float16* src = in + (((size_t)n * 122 + (r + dy)) * 122 + (c + dx)) * 16;
    const h8 v0 = *(const h8*)src;
    const h8 v1 = *(const h8*)(src + 8);
    #pragma unroll
    for (int ci = 0; ci < 8; ++ci) {
      acc += q8((float)v0[ci] * invS) * (int)w4q[t * 16 + ci];
      acc += q8((float)v1[ci] * invS) * (int)w4q[t * 16 + 8 + ci];
    }
  }
  const float y = (float)acc * s + b4[0];
  out[pix] = 1.0f / (1.0f + expf(-y));
}

extern "C" void kernel_launch(void* const* d_in, const int* in_sizes, int n_in,
                              void* d_out, int out_size, void* d_ws, size_t ws_size,
                              hipStream_t stream) {
  (void)in_sizes; (void)n_in; (void)out_size; (void)ws_size;
  const float* x  = (const float*)d_in[0];
  const float* w1 = (const float*)d_in[1];
  const float* b1 = (const float*)d_in[2];
  const float* w2 = (const float*)d_in[3];
  const float* b2 = (const float*)d_in[4];
  const float* w3 = (const float*)d_in[5];
  const float* b3 = (const float*)d_in[6];
  const float* w4 = (const float*)d_in[7];
  const float* b4 = (const float*)d_in[8];

  char*        ws  = (char*)d_ws;
  unsigned*    hdr = (unsigned*)(ws + OFF_HDR);
  signed char* w1q = (signed char*)(ws + OFF_W1Q);
  signed char* w4q = (signed char*)(ws + OFF_W4Q);
  int*         w2p = (int*)(ws + OFF_W2P);
  int*         w3p = (int*)(ws + OFF_W3P);
  _Float16*    A1  = (_Float16*)(ws + OFF_A1);
  _Float16*    A2  = (_Float16*)(ws + OFF_A2);
  _Float16*    A3  = (_Float16*)(ws + OFF_A3);

  qat_init<<<1, 64, 0, stream>>>(hdr);
  qat_absmax<<<4096, 256, 0, stream>>>(x, (long long)NBATCH * 128 * 128, hdr + 0);
  qat_prep<<<1, 256, 0, stream>>>(w1, w2, w3, w4, hdr, w1q, w2p, w3p, w4q);

  {
    const long long total = (long long)NBATCH * 126 * 126;
    qat_conv1<<<(int)((total + 255) / 256), 256, 0, stream>>>(x, A1, w1q, b1, hdr);
  }
  qat_conv3x3_i8wmma<<<NBATCH * 8 * 8, 256, 0, stream>>>(
      A1, A2, w2p, b2, hdr,  1,  9,  2,
      126, 126, 124, 124,  8,  8);
  qat_conv3x3_i8wmma<<<NBATCH * 8 * 8, 256, 0, stream>>>(
      A2, A3, w3p, b3, hdr, 2, 10, 3,
      124, 124, 122, 122, 8, 8);
  {
    const long long total = (long long)NBATCH * 121 * 121;
    qat_conv4<<<(int)((total + 255) / 256), 256, 0, stream>>>(A3, (float*)d_out, w4q, b4, hdr);
  }
}
